// SRUModel_51376398795434
// MI455X (gfx1250) — matrix operand order repaired and checked by reading, not compiled, not run on hardware
//
#include <hip/hip_runtime.h>
#include <cstdint>
#include <cstddef>

#define LSEQ  512
#define BATCH 128
#define DIN   128
#define HID   512
#define OUTN  5

typedef __attribute__((ext_vector_type(16))) __bf16 v16bf;
typedef __attribute__((ext_vector_type(8)))  __bf16 v8bf;
typedef __attribute__((ext_vector_type(8)))  float  v8f;

__device__ __forceinline__ unsigned short f32_to_bf16(float f) {
    union { float f; unsigned int u; } x; x.f = f;
    unsigned int r = x.u + 0x7fffu + ((x.u >> 16) & 1u);
    return (unsigned short)(r >> 16);
}

__device__ __forceinline__ float sigmoidf(float z) {
    return 1.0f / (1.0f + __expf(-z));
}

// one-off fp32 -> bf16 (RNE) conversion, grid-stride
__global__ void cvt_f32_to_bf16(const float* __restrict__ in,
                                unsigned short* __restrict__ out, int n) {
    int i = blockIdx.x * blockDim.x + threadIdx.x;
    int stride = gridDim.x * blockDim.x;
    for (; i < n; i += stride) out[i] = f32_to_bf16(in[i]);
}

// Fused SRU layer: per-wave 16(batch) x 16(hidden) tile, bf16-WMMA GEMM fused
// with the sequential recurrence; U never touches memory.
// A tiles are double-buffered in LDS via CDNA5 async copies
// (global_load_async_to_lds_b128 + s_wait_asynccnt): tile l+1 streams in while
// tile l feeds the WMMA pipe, removing global latency from the serial
// per-timestep critical path.
//   Abf: (LSEQ, BATCH, K) bf16    Wbf: (K, 3*HID) bf16
//   vc, bc: (2*HID) f32 gate params
//   out_seq: (LSEQ, BATCH, HID) bf16   (layer 0, feeds next layer's WMMA)
//   out_last: (BATCH, HID) f32         (layer 1, only h2[L-1] is needed)
template<int K, bool LAST_ONLY>
__global__ __launch_bounds__(32)
void sru_layer_wmma(const unsigned short* __restrict__ Abf,
                    const unsigned short* __restrict__ Wbf,
                    const float* __restrict__ vc,
                    const float* __restrict__ bc,
                    unsigned short* __restrict__ out_seq,
                    float* __restrict__ out_last)
{
    constexpr int NK    = K / 32;        // K-steps per timestep GEMM
    constexpr int NT    = 3 * NK;        // B tiles (u0,u1,u2 per K-step)
    constexpr int ROWB  = K * 2;         // A tile row bytes in global (power of 2)
    constexpr int PITCH = ROWB + 32;     // padded LDS row pitch (bank decorrelation)
    constexpr int NCH   = (16 * ROWB) / 512;  // async b128 issues per A tile
    constexpr int ABUF  = 16 * PITCH;    // one LDS A buffer, bytes

    __shared__ __align__(32) unsigned short smemB[NT * 32 * 16];
    __shared__ __align__(16) unsigned char  smemA[2 * ABUF];

    const int lane  = threadIdx.x;
    const int nl    = lane & 15;     // column within tile / A row index
    const int hi    = lane >> 4;     // lane half selects K sub-range
    const int btile = blockIdx.x;
    const int htile = blockIdx.y;
    const int hcol  = htile * 16 + nl;

    // ---- stage loop-invariant B tiles into LDS in per-lane WMMA-B layout ----
    // B 32x16 bf16: lanes 0-15 hold K=0..15 of their column, lanes 16-31 hold
    // K=16..31. Each lane's 16 bf16 contiguous -> 2x ds_load_b128 per tile.
    for (int ks = 0; ks < NK; ++ks) {
        for (int s = 0; s < 3; ++s) {
            const int t = ks * 3 + s;
            unsigned short* dst = &smemB[(t * 32 + lane) * 16];
            const unsigned short* src =
                Wbf + (size_t)(ks * 32 + hi * 16) * (3 * HID) + (size_t)s * HID + hcol;
            #pragma unroll
            for (int j = 0; j < 16; ++j) dst[j] = src[(size_t)j * (3 * HID)];
        }
    }
    __syncthreads();

    // LDS byte offset of the A staging area (flat LDS-aperture low 32 bits)
    const unsigned abase = (unsigned)(uintptr_t)&smemA[0];

    // async-copy one timestep's contiguous 16-row A tile into LDS buffer `buf`,
    // de-interleaving rows onto the padded pitch
    auto issue_tile = [&](int l, int buf) {
        const unsigned char* gbase =
            (const unsigned char*)(Abf + ((size_t)l * BATCH + btile * 16) * K);
        const unsigned lbase = abase + (unsigned)(buf * ABUF);
        #pragma unroll
        for (int cch = 0; cch < NCH; ++cch) {
            const unsigned t      = (unsigned)(cch * 512 + lane * 16);
            const unsigned row    = t / (unsigned)ROWB;   // const-folded shifts
            const unsigned within = t % (unsigned)ROWB;
            const unsigned ldsoff = lbase + row * (unsigned)PITCH + within;
            const unsigned long long ga = (unsigned long long)(uintptr_t)(gbase + t);
            asm volatile("global_load_async_to_lds_b128 %0, %1, off"
                         :: "v"(ldsoff), "v"(ga) : "memory");
        }
    };

    // per-lane gate parameters (depend only on hidden column)
    const float vf = vc[hcol], vr = vc[HID + hcol];
    const float bf = bc[hcol], br = bc[HID + hcol];

    v8f c = {};                              // recurrent cell state in C/D layout
    const int brow0 = btile * 16 + hi * 8;   // first C/D row this lane owns

    issue_tile(0, 0);                        // prologue: stage timestep 0

    for (int l = 0; l < LSEQ; ++l) {
        const int cur = l & 1;
        if (l + 1 < LSEQ) {
            issue_tile(l + 1, cur ^ 1);      // stream next tile behind compute
            // in-order completion: <=NCH outstanding => tile l fully in LDS
            asm volatile("s_wait_asynccnt %0" :: "n"(NCH) : "memory");
        } else {
            asm volatile("s_wait_asynccnt 0" ::: "memory");
        }

        const unsigned char* arow = smemA + cur * ABUF + nl * PITCH;

        v8f acc0 = {}, acc1 = {}, acc2 = {};
        #pragma unroll
        for (int ks = 0; ks < NK; ++ks) {
            // A 16x32 bf16 layout: lanes 0-15: K 0..15; 16-31: K 16..31
            const int kb = ks * 64 + hi * 32;            // byte offset of low chunk
            v8bf alo = *(const v8bf*)(arow + kb);
            v8bf ahi = *(const v8bf*)(arow + kb + 16);
            v16bf a;
            #pragma unroll
            for (int e = 0; e < 8; ++e) { a[e] = alo[e]; a[8 + e] = ahi[e]; }

            const v16bf b0 = *(const v16bf*)&smemB[((ks * 3 + 0) * 32 + lane) * 16];
            const v16bf b1 = *(const v16bf*)&smemB[((ks * 3 + 1) * 32 + lane) * 16];
            const v16bf b2 = *(const v16bf*)&smemB[((ks * 3 + 2) * 32 + lane) * 16];

            acc0 = __builtin_amdgcn_wmma_f32_16x16x32_bf16(false, a, false, b0,
                                                           (short)0, acc0, false, false);
            acc1 = __builtin_amdgcn_wmma_f32_16x16x32_bf16(false, a, false, b1,
                                                           (short)0, acc1, false, false);
            acc2 = __builtin_amdgcn_wmma_f32_16x16x32_bf16(false, a, false, b2,
                                                           (short)0, acc2, false, false);
        }

        // ---- recurrence directly on WMMA accumulator layout ----
        #pragma unroll
        for (int r = 0; r < 8; ++r) {
            const float u0 = acc0[r], u1 = acc1[r], u2 = acc2[r];
            const float cp = c[r];
            const float fg = sigmoidf(u1 + vf * cp + bf);
            const float rg = sigmoidf(u2 + vr * cp + br);
            const float cn = fg * cp + (1.0f - fg) * u0;
            c[r] = cn;
            const float ho = rg * cn;
            const int brow = brow0 + r;
            if (!LAST_ONLY) {
                out_seq[((size_t)l * BATCH + brow) * HID + hcol] = f32_to_bf16(ho);
            } else if (l == LSEQ - 1) {
                out_last[(size_t)brow * HID + hcol] = ho;
            }
        }
    }
}

// tiny FC head: out[b][o] = h2_last[b] . fc_w[o] + fc_b[o]
__global__ void fc_kernel(const float* __restrict__ h2, const float* __restrict__ fw,
                          const float* __restrict__ fb, float* __restrict__ out) {
    int idx = blockIdx.x * blockDim.x + threadIdx.x;
    if (idx >= BATCH * OUTN) return;
    const int b = idx / OUTN, o = idx % OUTN;
    const float* hv = h2 + (size_t)b * HID;
    const float* wv = fw + (size_t)o * HID;
    float s = 0.0f;
    #pragma unroll 4
    for (int h = 0; h < HID; ++h) s += hv[h] * wv[h];
    out[idx] = s + fb[o];
}

extern "C" void kernel_launch(void* const* d_in, const int* in_sizes, int n_in,
                              void* d_out, int out_size, void* d_ws, size_t ws_size,
                              hipStream_t stream) {
    const float* x   = (const float*)d_in[0];
    const float* W0  = (const float*)d_in[1];
    const float* v0  = (const float*)d_in[2];
    const float* b0  = (const float*)d_in[3];
    const float* W1  = (const float*)d_in[4];
    const float* v1  = (const float*)d_in[5];
    const float* b1  = (const float*)d_in[6];
    const float* fcw = (const float*)d_in[7];
    const float* fcb = (const float*)d_in[8];
    float* out = (float*)d_out;

    // workspace layout (bytes)
    char* ws = (char*)d_ws;
    unsigned short* xb  = (unsigned short*)(ws);              // x bf16:  16,777,216
    unsigned short* w0b = (unsigned short*)(ws + 16777216);   // W0 bf16:    393,216
    unsigned short* w1b = (unsigned short*)(ws + 17170432);   // W1 bf16:  1,572,864
    unsigned short* h1b = (unsigned short*)(ws + 18743296);   // h1 bf16: 67,108,864
    float*          h2l = (float*)(ws + 85852160);            // h2[L-1]:    262,144

    cvt_f32_to_bf16<<<2048, 256, 0, stream>>>(x,  xb,  LSEQ * BATCH * DIN);
    cvt_f32_to_bf16<<<192,  256, 0, stream>>>(W0, w0b, DIN * 3 * HID);
    cvt_f32_to_bf16<<<768,  256, 0, stream>>>(W1, w1b, HID * 3 * HID);

    dim3 grid(BATCH / 16, HID / 16);   // 8 x 32 = 256 single-wave tiles
    sru_layer_wmma<DIN, false><<<grid, 32, 0, stream>>>(xb,  w0b, v0, b0, h1b, nullptr);
    sru_layer_wmma<HID, true ><<<grid, 32, 0, stream>>>(h1b, w1b, v1, b1, nullptr, h2l);

    fc_kernel<<<(BATCH * OUTN + 127) / 128, 128, 0, stream>>>(h2l, fcw, fcb, out);
}
